// DeformConv2D_575525617718
// MI455X (gfx1250) — matrix operand order repaired and checked by reading, not compiled, not run on hardware
//
#include <hip/hip_runtime.h>
#include <hip/hip_bf16.h>

typedef __attribute__((ext_vector_type(16))) _Float16 v16h;
typedef __attribute__((ext_vector_type(8)))  _Float16 v8h;
typedef __attribute__((ext_vector_type(8)))  float    v8f;

#define KSZ      3
#define TAPS     9          // KSZ*KSZ
#define C_IN     128
#define C_OUT    128
#define HH       64
#define WW       64
#define NB       4
#define PLANE    (HH*WW)    // 4096
#define KTOT     (C_IN*TAPS)  // 1152
#define KCHUNK   32
#define NKC      (KTOT/KCHUNK) // 36
#define PIXBLK   64
#define THREADS  256

struct Meta {
    float    g[4];     // bilinear weights (zeroed if sample OOB)
    unsigned off01;    // two packed u16 plane offsets
    unsigned off23;
};

// ---------------------------------------------------------------------------
// Kernel 1: weight f32 -> f16, row-major [oc][c*9 + kh*3 + kw]  (= A matrix)
// ---------------------------------------------------------------------------
__global__ void wcvt_kernel(const float* __restrict__ w,
                            _Float16* __restrict__ w16, int n) {
    int i = blockIdx.x * blockDim.x + threadIdx.x;
    if (i < n) w16[i] = (_Float16)w[i];
}

// ---------------------------------------------------------------------------
// Kernel 2: fused bilinear sampling + implicit GEMM via WMMA
//   block: 64 pixels (one image row, one batch) x all 128 output channels
//   wave w (0..7): rows oc = 16w..16w+15, 4 accumulator tiles of 16x16
// ---------------------------------------------------------------------------
__global__ __launch_bounds__(THREADS)
void deform_wmma_kernel(const float* __restrict__ x,
                        const float* __restrict__ offset,
                        const _Float16* __restrict__ w16,
                        float* __restrict__ out) {
    __shared__ Meta smeta[PIXBLK * TAPS];                         // 13824 B
    __shared__ __align__(16) _Float16 sB[2][PIXBLK][KCHUNK + 8];  // 80B rows, 10240 B

    const int p0   = blockIdx.x * PIXBLK;     // global pixel base
    const int b    = p0 >> 12;                // batch
    const int row  = (p0 & (PLANE - 1)) >> 6; // image row (0..63)
    const float* __restrict__ xb   = x + (size_t)b * C_IN * PLANE;
    const float* __restrict__ offb = offset + (size_t)b * (2 * TAPS) * PLANE;

    // ---------------- Phase A: per-(pixel, tap) bilinear metadata ----------
    for (int e = threadIdx.x; e < PIXBLK * TAPS; e += THREADS) {
        const int j = e / TAPS;       // pixel column
        const int t = e - j * TAPS;   // tap (kh*3+kw)
        const float offx = offb[(2 * t    ) * PLANE + row * WW + j];
        const float offy = offb[(2 * t + 1) * PLANE + row * WW + j];
        float px = (float)(row + (t / 3)) + offx;   // row+1 + (t/3-1) + off
        float py = (float)(j   + (t % 3)) + offy;
        const float fx = floorf(px), fy = floorf(py);
        int qltx = (int)fx;     qltx = qltx < 0 ? 0 : (qltx > 65 ? 65 : qltx);
        int qlty = (int)fy;     qlty = qlty < 0 ? 0 : (qlty > 65 ? 65 : qlty);
        int qrbx = (int)fx + 1; qrbx = qrbx < 0 ? 0 : (qrbx > 65 ? 65 : qrbx);
        int qrby = (int)fy + 1; qrby = qrby < 0 ? 0 : (qrby > 65 ? 65 : qrby);
        // border snap (reference's mask trick), then clamp
        if (px < 1.f || px > 64.f) px = fx;
        if (py < 1.f || py > 64.f) py = fy;
        px = fminf(fmaxf(px, 0.f), 65.f);
        py = fminf(fmaxf(py, 0.f), 65.f);
        const float glx = 1.f + ((float)qltx - px);
        const float grx = 1.f - ((float)qrbx - px);
        const float gly = 1.f + ((float)qlty - py);
        const float gry = 1.f - ((float)qrby - py);
        float g0 = glx * gly, g1 = grx * gry, g2 = glx * gry, g3 = grx * gly;
        // fold zero-padding into weights; clamp offsets into the real plane
        unsigned o0, o1, o2, o3;
        {
            bool v;
            v = (qltx >= 1) & (qltx <= 64) & (qlty >= 1) & (qlty <= 64);
            o0 = v ? (unsigned)((qltx - 1) * WW + (qlty - 1)) : 0u; if (!v) g0 = 0.f;
            v = (qrbx >= 1) & (qrbx <= 64) & (qrby >= 1) & (qrby <= 64);
            o1 = v ? (unsigned)((qrbx - 1) * WW + (qrby - 1)) : 0u; if (!v) g1 = 0.f;
            v = (qltx >= 1) & (qltx <= 64) & (qrby >= 1) & (qrby <= 64);
            o2 = v ? (unsigned)((qltx - 1) * WW + (qrby - 1)) : 0u; if (!v) g2 = 0.f;
            v = (qrbx >= 1) & (qrbx <= 64) & (qlty >= 1) & (qlty <= 64);
            o3 = v ? (unsigned)((qrbx - 1) * WW + (qlty - 1)) : 0u; if (!v) g3 = 0.f;
        }
        Meta m;
        m.g[0] = g0; m.g[1] = g1; m.g[2] = g2; m.g[3] = g3;
        m.off01 = o0 | (o1 << 16);
        m.off23 = o2 | (o3 << 16);
        smeta[e] = m;
    }
    __syncthreads();

    // ---------------- Phase B: K-chunked WMMA GEMM -------------------------
    const int wave  = threadIdx.x >> 5;      // 0..7 -> oc tile
    const int lane  = threadIdx.x & 31;
    const int lhalf = lane >> 4;             // 0/1
    const int lmod  = lane & 15;
    const int ocb   = wave * 16;

    // A-fragment source: row-major weights. Lane holds M=lmod; halves
    // [0..8) at col kb + lhalf*8, halves [8..16) at col kb + lhalf*8 + 16.
    const _Float16* __restrict__ arow =
        w16 + (size_t)(ocb + lmod) * KTOT + lhalf * 8;

    // B-fill assignment: fixed pixel, 8 consecutive k values per thread
    const int fillp = threadIdx.x & 63;      // pixel 0..63
    const int fillg = threadIdx.x >> 6;      // k-subgroup 0..3

    v8f acc[4] = {};

    for (int kc = 0; kc < NKC; ++kc) {
        const int kb  = kc * KCHUNK;
        const int buf = kc & 1;

        // ---- cooperative sampling of the 32x64 f16 B tile into LDS ----
        {
            int k0 = kb + fillg * 8;
            int c  = k0 / TAPS;
            int t  = k0 - c * TAPS;
            v8h vals;
            const Meta* __restrict__ mp = &smeta[fillp * TAPS];
#pragma unroll
            for (int u = 0; u < 8; ++u) {
                const Meta m = mp[t];
                const float* __restrict__ xp = xb + (size_t)c * PLANE;
                float v = m.g[0] * xp[m.off01 & 0xffffu]
                        + m.g[1] * xp[m.off01 >> 16]
                        + m.g[2] * xp[m.off23 & 0xffffu]
                        + m.g[3] * xp[m.off23 >> 16];
                vals[u] = (_Float16)v;
                if (++t == TAPS) { t = 0; ++c; }
            }
            *(v8h*)(&sB[buf][fillp][fillg * 8]) = vals;   // 16B ds_store
        }
        __syncthreads();   // double-buffered: single barrier per chunk

        // ---- A fragment (two 16B global loads, documented 16-bit A layout)
        v16h afrag;
        {
            const _Float16* ap = arow + kb;
            v8h alo = *(const v8h*)ap;
            v8h ahi = *(const v8h*)(ap + 16);
#pragma unroll
            for (int u = 0; u < 8; ++u) { afrag[u] = alo[u]; afrag[u + 8] = ahi[u]; }
        }

        // ---- 4 N-subtiles: B fragment (K at lhalf*8 and lhalf*8+16, 8 each) + WMMA
#pragma unroll
        for (int n = 0; n < 4; ++n) {
            const _Float16* bp = &sB[buf][n * 16 + lmod][lhalf * 8];
            v8h blo = *(const v8h*)bp;
            v8h bhi = *(const v8h*)(bp + 16);
            v16h bfrag;
#pragma unroll
            for (int u = 0; u < 8; ++u) { bfrag[u] = blo[u]; bfrag[u + 8] = bhi[u]; }
            acc[n] = __builtin_amdgcn_wmma_f32_16x16x32_f16(
                /*neg_a=*/false, afrag, /*neg_b=*/false, bfrag,
                /*c_mod=*/(short)0, acc[n], /*reuse_a=*/false, /*reuse_b=*/false);
        }
    }

    // ---------------- Epilogue: scatter C (documented f32 C/D layout) ------
    float* __restrict__ outb = out + (size_t)b * C_OUT * PLANE + row * WW;
#pragma unroll
    for (int n = 0; n < 4; ++n) {
        const int j = n * 16 + lmod;
#pragma unroll
        for (int r = 0; r < 8; ++r) {
            const int oc = ocb + r + 8 * lhalf;
            outb[(size_t)oc * PLANE + j] = acc[n][r];
        }
    }
}

// ---------------------------------------------------------------------------
extern "C" void kernel_launch(void* const* d_in, const int* in_sizes, int n_in,
                              void* d_out, int out_size, void* d_ws, size_t ws_size,
                              hipStream_t stream) {
    const float* x      = (const float*)d_in[0];  // (4,128,64,64)
    const float* offset = (const float*)d_in[1];  // (4,18,64,64)
    const float* weight = (const float*)d_in[2];  // (128,128,3,3)
    float* out          = (float*)d_out;          // (4,128,64,64)

    _Float16* w16 = (_Float16*)d_ws;              // 128*1152 f16 = 288 KB
    const int nw = C_OUT * KTOT;
    wcvt_kernel<<<(nw + 255) / 256, 256, 0, stream>>>(weight, w16, nw);

    const int nblocks = (NB * PLANE) / PIXBLK;    // 256
    deform_wmma_kernel<<<nblocks, THREADS, 0, stream>>>(x, offset, w16, out);
}
